// ResidualVQ_46935402611149
// MI455X (gfx1250) — matrix operand order repaired and checked by reading, not compiled, not run on hardware
//
#include <hip/hip_runtime.h>

#define B_N 65536
#define D_N 512
#define K_N 1024
#define Q_N 8

typedef __bf16 bf16_t;
typedef bf16_t v8bf  __attribute__((ext_vector_type(8)));
typedef bf16_t v16bf __attribute__((ext_vector_type(16)));
typedef float  v8f   __attribute__((ext_vector_type(8)));
typedef int    v4i_t __attribute__((ext_vector_type(4)));

typedef __attribute__((address_space(1))) v4i_t* gv4i_p;   // global int4*
typedef __attribute__((address_space(3))) v4i_t* lv4i_p;   // LDS int4*

struct alignas(8) bf4 { bf16_t a, b, c, d; };

#if __has_builtin(__builtin_amdgcn_global_load_async_to_lds_b128)
#define HAVE_ASYNC_LDS 1
#else
#define HAVE_ASYNC_LDS 0
#endif

// 16B-granular XOR swizzle of the within-row byte offset so 16 same-column
// reads of a WMMA tile (1024-B row stride) hit distinct LDS bank groups.
__device__ __forceinline__ int swz(int row, int byteoff) {
    return byteoff ^ (((row & 3) << 6) | (((row >> 2) & 3) << 4));
}

// 16-byte global -> LDS copy: async on gfx1250, sync fallback otherwise.
__device__ __forceinline__ void cp16(const bf16_t* g, bf16_t* l) {
#if HAVE_ASYNC_LDS
    __builtin_amdgcn_global_load_async_to_lds_b128((gv4i_p)(g), (lv4i_p)(l), 0, 0);
#else
    *(v8bf*)l = *(const v8bf*)g;
#endif
}

__device__ __forceinline__ void wait_async0() {
#if HAVE_ASYNC_LDS
#if __has_builtin(__builtin_amdgcn_s_wait_asynccnt)
    __builtin_amdgcn_s_wait_asynccnt(0);
#else
    asm volatile("s_wait_asynccnt 0x0" ::: "memory");
#endif
#endif
}

// ---------------------------------------------------------------------------
// Init: codebooks f32 -> bf16, and exact f32 ||c_k||^2 per code row.
// ---------------------------------------------------------------------------
__global__ __launch_bounds__(128) void rvq_init(const float* __restrict__ cb,
                                                bf16_t* __restrict__ cbbf,
                                                float* __restrict__ cnorm) {
    __shared__ float red[4];
    const int row = blockIdx.x;
    const int tid = threadIdx.x;
    const float4* src = (const float4*)(cb + (size_t)row * D_N);
    float4 v = src[tid];
    bf4 o = { (bf16_t)v.x, (bf16_t)v.y, (bf16_t)v.z, (bf16_t)v.w };
    *(bf4*)(cbbf + (size_t)row * D_N + tid * 4) = o;
    float s = v.x * v.x + v.y * v.y + v.z * v.z + v.w * v.w;
    #pragma unroll
    for (int off = 16; off >= 1; off >>= 1) s += __shfl_xor(s, off, 32);
    if ((tid & 31) == 0) red[tid >> 5] = s;
    __syncthreads();
    if (tid == 0) cnorm[row] = red[0] + red[1] + red[2] + red[3];
}

// ---------------------------------------------------------------------------
// Fused distance GEMM + argmin. 4 wave32 / WG; A (64 rows) staged via LDS into
// registers; B (codebook) double-buffered in the same LDS via async copies.
// ---------------------------------------------------------------------------
#define AM_WAVES   4
#define AM_THREADS (AM_WAVES * 32)
#define AM_ROWS    (AM_WAVES * 16)        // 64 rows
#define B_CHUNK_BYTES (16 * D_N * 2)      // 16 codes x 512 bf16 = 16KB

__global__ __launch_bounds__(AM_THREADS) void rvq_argmin(
    const float* __restrict__ src,      // residual (B,D) f32
    const bf16_t* __restrict__ cbq,     // stage codebook (K,D) bf16
    const float* __restrict__ cnq,      // stage code norms (K) f32
    int* __restrict__ idx_ws,           // (B) winner index
    float* __restrict__ idx_out,        // d_out indices region, stride Q
    int q) {
    __shared__ __align__(16) char lds_raw[AM_ROWS * D_N * 2];   // 64KB, reused
    const int tid  = threadIdx.x;
    const int lane = tid & 31;
    const int wave = tid >> 5;
    const long rowBase = (long)blockIdx.x * AM_ROWS;

    // Phase 1: stage 64 residual rows into LDS as bf16 (swizzled).
    {
        bf16_t* smemA = (bf16_t*)lds_raw;
        const float4* sv = (const float4*)(src + rowBase * D_N);
        #pragma unroll 4
        for (int i = 0; i < (AM_ROWS * D_N / 4) / AM_THREADS; ++i) {
            int id = tid + i * AM_THREADS;
            int r  = id >> 7;
            int c4 = id & 127;
            float4 v = sv[id];
            bf4 o = { (bf16_t)v.x, (bf16_t)v.y, (bf16_t)v.z, (bf16_t)v.w };
            *(bf4*)((char*)(smemA + r * D_N) + swz(r, c4 * 8)) = o;
        }
    }
    __syncthreads();

    // Phase 2: explicit A-fragment preload into registers (frees LDS for B).
    // A layout (ISA): lanes l / l+16 hold same M=l&15; K halves 0-7,16-23 /
    // 8-15,24-31 of each 32-wide chunk.
    const int arow  = wave * 16 + (lane & 15);
    const int aoff0 = (lane < 16) ? 0 : 16;   // bytes
    v16bf afrag[16];
    {
        const char* arowp = (const char*)(lds_raw) + arow * (D_N * 2);
        #pragma unroll
        for (int dc = 0; dc < D_N / 32; ++dc) {
            int off = dc * 64 + aoff0;
            v8bf a0 = *(const v8bf*)(arowp + swz(arow, off));
            v8bf a1 = *(const v8bf*)(arowp + swz(arow, off + 32));
            afrag[dc] = __builtin_shufflevector(a0, a1, 0, 1, 2, 3, 4, 5, 6, 7,
                                                8, 9, 10, 11, 12, 13, 14, 15);
        }
    }
    __syncthreads();   // all waves done reading A; LDS may be overwritten

    // Phase 3: kc loop with double-buffered async codebook staging.
    const int bn    = lane & 15;              // B: N = lane&15
    const int bbyte = (lane < 16) ? 0 : 16;   // K 0-7,16-23 / 8-15,24-31 (bytes)

    auto issue_chunk = [&](int kcn, char* dstbuf) {
        const bf16_t* gbase = cbq + (size_t)kcn * 16 * D_N;
        #pragma unroll
        for (int it = 0; it < 8; ++it) {
            int e  = tid + it * AM_THREADS;   // 0..1023 16B-packets
            int n  = e >> 6;                  // code row in chunk
            int ro = (e & 63) * 16;           // byte offset within row
            cp16(gbase + (size_t)e * 8,
                 (bf16_t*)(dstbuf + n * 1024 + swz(n, ro)));
        }
    };

    float minv[8];
    int   mini[8];
    #pragma unroll
    for (int v = 0; v < 8; ++v) { minv[v] = 3.402823466e38f; mini[v] = 0; }

    issue_chunk(0, lds_raw);

    for (int kc = 0; kc < K_N / 16; ++kc) {
        char* cur = lds_raw + (kc & 1) * B_CHUNK_BYTES;
        char* nxt = lds_raw + ((kc + 1) & 1) * B_CHUNK_BYTES;

        wait_async0();       // my async copies for `cur` are complete
        __syncthreads();     // everyone's copies for `cur` are complete
        if (kc + 1 < K_N / 16) issue_chunk(kc + 1, nxt);  // overlaps compute

        float cn = cnq[kc * 16 + bn];
        const char* brow = cur + bn * 1024;

        // Two accumulation chains (even/odd dc) for matrix-pipe ILP.
        v8f acc0 = {0.f, 0.f, 0.f, 0.f, 0.f, 0.f, 0.f, 0.f};
        v8f acc1 = {0.f, 0.f, 0.f, 0.f, 0.f, 0.f, 0.f, 0.f};
        #pragma unroll
        for (int dc = 0; dc < D_N / 32; dc += 2) {
            int off0 = dc * 64 + bbyte;
            v8bf b00 = *(const v8bf*)(brow + swz(bn, off0));
            v8bf b01 = *(const v8bf*)(brow + swz(bn, off0 + 32));
            v16bf b0 = __builtin_shufflevector(b00, b01, 0, 1, 2, 3, 4, 5, 6, 7,
                                               8, 9, 10, 11, 12, 13, 14, 15);
            int off1 = (dc + 1) * 64 + bbyte;
            v8bf b10 = *(const v8bf*)(brow + swz(bn, off1));
            v8bf b11 = *(const v8bf*)(brow + swz(bn, off1 + 32));
            v16bf b1 = __builtin_shufflevector(b10, b11, 0, 1, 2, 3, 4, 5, 6, 7,
                                               8, 9, 10, 11, 12, 13, 14, 15);
            acc0 = __builtin_amdgcn_wmma_f32_16x16x32_bf16(
                false, afrag[dc], false, b0, (short)0, acc0, false, false);
            acc1 = __builtin_amdgcn_wmma_f32_16x16x32_bf16(
                false, afrag[dc + 1], false, b1, (short)0, acc1, false, false);
        }

        int kg = kc * 16 + bn;
        #pragma unroll
        for (int v = 0; v < 8; ++v) {
            float dv = fmaf(-2.0f, acc0[v] + acc1[v], cn);
            if (dv < minv[v]) { minv[v] = dv; mini[v] = kg; }  // earliest k ties
        }
        __syncthreads();   // all waves done reading `cur` before it is refilled
    }

    // Cross-lane (column) reduction in each 16-lane half; first-min ties.
    #pragma unroll
    for (int off = 8; off >= 1; off >>= 1) {
        #pragma unroll
        for (int v = 0; v < 8; ++v) {
            float ov = __shfl_xor(minv[v], off, 32);
            int   oi = __shfl_xor(mini[v], off, 32);
            if (ov < minv[v] || (ov == minv[v] && oi < mini[v])) {
                minv[v] = ov; mini[v] = oi;
            }
        }
    }
    if ((lane & 15) == 0) {
        long r0 = rowBase + wave * 16 + (lane >> 4) * 8;
        #pragma unroll
        for (int v = 0; v < 8; ++v) {
            idx_ws[r0 + v] = mini[v];
            idx_out[(r0 + v) * Q_N + q] = (float)mini[v];
        }
    }
}

// ---------------------------------------------------------------------------
// Exact f32 residual update + per-row loss partial.
// ---------------------------------------------------------------------------
__global__ __launch_bounds__(128) void rvq_update(const float* __restrict__ src,
                                                  float* __restrict__ res,
                                                  const float* __restrict__ cbq,
                                                  const int* __restrict__ idx_ws,
                                                  float* __restrict__ partial) {
    __shared__ float red[4];
    const long b   = blockIdx.x;
    const int  tid = threadIdx.x;
    const int  k   = idx_ws[b];
    const float4* rp = (const float4*)(src + b * D_N);
    const float4* cp = (const float4*)(cbq + (size_t)k * D_N);
    float4 r = rp[tid];
    float4 c = cp[tid];
    float4 n = make_float4(r.x - c.x, r.y - c.y, r.z - c.z, r.w - c.w);
    ((float4*)(res + b * D_N))[tid] = n;
    float s = n.x * n.x + n.y * n.y + n.z * n.z + n.w * n.w;  // (q-r)^2 == n^2
    #pragma unroll
    for (int off = 16; off >= 1; off >>= 1) s += __shfl_xor(s, off, 32);
    if ((tid & 31) == 0) red[tid >> 5] = s;
    __syncthreads();
    if (tid == 0) partial[b] = red[0] + red[1] + red[2] + red[3];
}

// Deterministic single-WG reduction of B per-row partials -> loss[q].
__global__ __launch_bounds__(1024) void rvq_reduce(const float* __restrict__ partial,
                                                   float* __restrict__ lossAcc,
                                                   int q) {
    __shared__ float red[32];
    const int tid = threadIdx.x;
    float s = 0.f;
    for (int i = tid; i < B_N; i += 1024) s += partial[i];
    #pragma unroll
    for (int off = 16; off >= 1; off >>= 1) s += __shfl_xor(s, off, 32);
    if ((tid & 31) == 0) red[tid >> 5] = s;
    __syncthreads();
    if (tid < 32) {
        float t = red[tid];
        #pragma unroll
        for (int off = 16; off >= 1; off >>= 1) t += __shfl_xor(t, off, 32);
        if (tid == 0) lossAcc[q] = t;
    }
}

// y_hat = y - final_residual; write losses and loss_vq.
__global__ __launch_bounds__(256) void rvq_final(const float* __restrict__ y,
                                                 const float* __restrict__ res,
                                                 float* __restrict__ out,
                                                 const float* __restrict__ lossAcc) {
    const long n4 = (long)B_N * D_N / 4;
    const long stride = (long)gridDim.x * blockDim.x;
    const float4* yv = (const float4*)y;
    const float4* rv = (const float4*)res;
    float4* ov = (float4*)out;
    for (long i = (long)blockIdx.x * blockDim.x + threadIdx.x; i < n4; i += stride) {
        float4 a = yv[i], b = rv[i];
        ov[i] = make_float4(a.x - b.x, a.y - b.y, a.z - b.z, a.w - b.w);
    }
    if (blockIdx.x == 0 && threadIdx.x == 0) {
        const double invN = 1.0 / ((double)B_N * (double)D_N);
        const long base = (long)B_N * D_N + (long)B_N * Q_N;
        float s = 0.f;
        for (int q = 0; q < Q_N; ++q) {
            float l = (float)((double)lossAcc[q] * invN);  // COMMITMENT_WEIGHT = 1
            out[base + 1 + q] = l;
            s += l;
        }
        out[base] = s / (float)Q_N;
    }
}

// ---------------------------------------------------------------------------
extern "C" void kernel_launch(void* const* d_in, const int* in_sizes, int n_in,
                              void* d_out, int out_size, void* d_ws, size_t ws_size,
                              hipStream_t stream) {
    (void)in_sizes; (void)n_in; (void)out_size; (void)ws_size;
    const float* y  = (const float*)d_in[0];                 // (B, D) f32
    const float* cb = (const float*)d_in[1];                 // (Q, K, D) f32
    float* out = (float*)d_out;

    char* p = (char*)d_ws;
    float*  res   = (float*)p;  p += (size_t)B_N * D_N * sizeof(float);
    bf16_t* cbbf  = (bf16_t*)p; p += (size_t)Q_N * K_N * D_N * sizeof(bf16_t);
    float*  cnorm = (float*)p;  p += (size_t)Q_N * K_N * sizeof(float);
    int*    idxw  = (int*)p;    p += (size_t)B_N * sizeof(int);
    float*  part  = (float*)p;  p += (size_t)B_N * sizeof(float);
    float*  lossA = (float*)p;

    rvq_init<<<Q_N * K_N, 128, 0, stream>>>(cb, cbbf, cnorm);

    for (int q = 0; q < Q_N; ++q) {
        const float* srcRes = (q == 0) ? y : res;
        rvq_argmin<<<B_N / AM_ROWS, AM_THREADS, 0, stream>>>(
            srcRes, cbbf + (size_t)q * K_N * D_N, cnorm + q * K_N,
            idxw, out + (size_t)B_N * D_N, q);
        rvq_update<<<B_N, 128, 0, stream>>>(
            srcRes, res, cb + (size_t)q * K_N * D_N, idxw, part);
        rvq_reduce<<<1, 1024, 0, stream>>>(part, lossA, q);
    }

    rvq_final<<<2048, 256, 0, stream>>>(y, res, out, lossA);
}
